// PF_LSTM_65919158059070
// MI455X (gfx1250) — matrix operand order repaired and checked by reading, not compiled, not run on hardware
//
#include <hip/hip_runtime.h>
#include <math.h>

typedef unsigned int u32;
typedef unsigned short u16;
typedef unsigned long long u64;

typedef __attribute__((ext_vector_type(16))) __bf16 v16bf;
typedef __attribute__((ext_vector_type(8)))  float  v8f;
typedef __attribute__((ext_vector_type(4)))  u32    v4u;
typedef __attribute__((ext_vector_type(8)))  int    v8i;
typedef __attribute__((ext_vector_type(4)))  int    v4i;

struct __align__(16) U4 { u32 a, b, c, d; };

union V16 { v16bf v; U4 q[2]; u16 us[16]; };

#define LS 2
#define BS 16
#define T_STEPS 16
#define DIN 32
#define NH 64
#define NP 2048
#define HALF_NP 1024
#define NT 16
#define N_WAVES 8

// ---- workspace byte offsets ----
#define WS_WB    0                       // 112 tiles * 1024 B bf16 B-operand layout
#define WS_BREP  114688                  // replicated bias: [2][256][8] f32 = 16384 B
#define WS_XA    131072                  // x in bf16 A-operand layout: T*32 lanes*16
#define WS_PPRE  147456                  // pre-resample particles: 4194304 f32
#define WS_Y     16924672                // Y: 16*2048 f32
#define WS_NEWW  17055744                // new_w: 2048 f32
#define WS_STATS 17063936                // neff per half
#define WS_YF    17064192                // Yf: 16*2048 f32

#define WCOPY_BYTES 131072               // weights + replicated bias, one TDM descriptor
#define WCOPY8      (WCOPY_BYTES / 8)    // 16384 8-byte elements (fits tile_dim0 u16)

// ---- d_out float offsets (tuple flattened in return order) ----
#define O_WMU 0
#define O_COV 16
#define O_PF  272
#define O_WF  (272 + 4194304)
#define O_L   (O_WF + 2048)
#define O_NIS (O_L + 1)

__device__ __forceinline__ u16 f2bf(float f) {
  union { float f; u32 u; } v; v.f = f;
  u32 r = v.u + 0x7FFFu + ((v.u >> 16) & 1u);
  return (u16)(r >> 16);
}
__device__ __forceinline__ float bf2f(u16 h) {
  union { u32 u; float f; } v; v.u = ((u32)h) << 16;
  return v.f;
}
// branchless activations: single v_exp_f32 + v_rcp_f32, no EXEC divergence
__device__ __forceinline__ float fsig(float x) {
  return __builtin_amdgcn_rcpf(1.0f + __builtin_amdgcn_exp2f(-1.44269504089f * x));
}
__device__ __forceinline__ float ftanh(float x) {
  return 2.0f * fsig(2.0f * x) - 1.0f;
}
__device__ __forceinline__ float gnoise(u32 s) {
  s ^= s >> 17; s *= 0xed5ad4bbu; s ^= s >> 11; s *= 0xac4c1b51u;
  s ^= s >> 15; s *= 0x31848babu; s ^= s >> 14;
  u32 s2 = s * 747796405u + 2891336453u; s2 ^= s2 >> 13; s2 *= 0x9E3779B1u; s2 ^= s2 >> 16;
  float u1 = ((s & 0xFFFFFFu) + 1.0f) * (1.0f / 16777217.0f);
  float u2 = (s2 & 0xFFFFFFu) * (1.0f / 16777216.0f);
  return sqrtf(-2.0f * logf(u1)) * __cosf(6.28318530718f * u2);
}

// =====================================================================
// Prep: fp32 weights/x -> bf16 WMMA operand layouts; bias replicated x8
// so it can be ds_loaded directly as a v8f WMMA C operand.
// =====================================================================
__global__ void prep_kernel(const float* __restrict__ x,
                            const float* __restrict__ Wih0, const float* __restrict__ Whh0,
                            const float* __restrict__ bih0, const float* __restrict__ bhh0,
                            const float* __restrict__ Wih1, const float* __restrict__ Whh1,
                            const float* __restrict__ bih1, const float* __restrict__ bhh1,
                            u16* __restrict__ wB, float* __restrict__ brep,
                            u16* __restrict__ xA) {
  int tid = blockIdx.x * blockDim.x + threadIdx.x;
  int nthr = gridDim.x * blockDim.x;
  // 112 tiles * 512 bf16: [0,16)=Wih0, [16,48)=Whh0, [48,80)=Wih1, [80,112)=Whh1
  for (int idx = tid; idx < 112 * 512; idx += nthr) {
    int t = idx >> 9, w = idx & 511;
    int lane = w >> 4, j = w & 15;
    const float* W; int Kin, nt, kc;
    if (t < 16)      { W = Wih0; Kin = 32; nt = t;           kc = 0; }
    else if (t < 48) { W = Whh0; Kin = 64; nt = (t - 16) >> 1; kc = (t - 16) & 1; }
    else if (t < 80) { W = Wih1; Kin = 64; nt = (t - 48) >> 1; kc = (t - 48) & 1; }
    else             { W = Whh1; Kin = 64; nt = (t - 80) >> 1; kc = (t - 80) & 1; }
    int col = nt * 16 + (lane & 15);
    int k   = kc * 32 + ((j < 8) ? (((lane >= 16) ? 8 : 0) + j) : (16 + ((lane >= 16) ? 8 : 0) + (j - 8)));
    wB[idx] = f2bf(W[col * Kin + k]);            // B[k][col] = W[col][k]
  }
  // replicated bias: [l][col][v] = bih_l[col] + bhh_l[col], v=0..7
  for (int idx = tid; idx < 2 * 256 * 8; idx += nthr) {
    int l = idx >> 11;
    int col = (idx >> 3) & 255;
    brep[idx] = l ? (bih1[col] + bhh1[col]) : (bih0[col] + bhh0[col]);
  }
  // x A-layout: [t][lane][j]
  for (int idx = tid; idx < T_STEPS * 512; idx += nthr) {
    int t = idx >> 9, w = idx & 511;
    int lane = w >> 4, j = w & 15;
    int m = lane & 15, off = (lane >= 16) ? 8 : 0;
    int k = (j < 8) ? (off + j) : (16 + off + (j - 8));
    xA[idx] = f2bf(x[(m * T_STEPS + t) * DIN + k]);
  }
}

// =====================================================================
// Main: one wave32 == one particle. 2-layer, 16-step LSTM via
// v_wmma_f32_16x16x32_bf16. Weights + replicated bias staged to LDS via
// TDM (tensor_load_to_lds), overlapped with particle-state init loads.
// Bias enters as a ds-loaded v8f C operand (no splat, no spills).
// =====================================================================
__global__ void __launch_bounds__(256)
pf_lstm_wmma_kernel(const u16* __restrict__ wBg,
                    const u16* __restrict__ xAg, const float* __restrict__ pin,
                    float* __restrict__ ppre,
                    const float* __restrict__ qp, const float* __restrict__ ep) {
  extern __shared__ U4 smem_u4[];
  char* smem = (char*)smem_u4;
  u16*   wlds = (u16*)smem;                 // 114688 B  (LDS offset 0)
  float* brep = (float*)(smem + 114688);    // 16384 B
  u16*   hlds = (u16*)(smem + 131072);      // 8 waves * 2 layers * 1024 u16 = 32768 B

  const int tid  = threadIdx.x;
  const int lane = tid & 31;
  const int wave = tid >> 5;
  const int p    = blockIdx.x * N_WAVES + wave;

  // ---- TDM: one 1-D descriptor DMAs weights + replicated bias into LDS ----
  if (wave == 0) {
    u64 ga = (u64)wBg;
    v4u g0;
    g0[0] = 1u;                                            // count=1, user mode
    g0[1] = 0u;                                            // lds_addr = 0 (wlds base)
    g0[2] = (u32)ga;                                       // global_addr[31:0]
    g0[3] = ((u32)(ga >> 32) & 0x01FFFFFFu) | (2u << 30);  // addr[56:32] | type=2
    v8i g1;
    g1[0] = 0x00030000;                                    // data_size = 3 (8 bytes)
    g1[1] = (int)((WCOPY8 & 0xFFFFu) << 16);               // tensor_dim0[15:0]
    g1[2] = (int)(((WCOPY8 >> 16) & 0xFFFFu) | (1u << 16));// dim0[31:16], tensor_dim1=1
    g1[3] = (int)((WCOPY8 & 0xFFFFu) << 16);               // tile_dim0 = WCOPY8
    g1[4] = 0;                                             // tile_dim1/2 = 0 (1-D)
    g1[5] = (int)WCOPY8;                                   // tensor_dim0_stride
    g1[6] = 0;
    g1[7] = 0;
    v4i z4 = {0, 0, 0, 0};
    v8i z8 = {0, 0, 0, 0, 0, 0, 0, 0};
    __builtin_amdgcn_tensor_load_to_lds(g0, g1, z4, z4, z8, 0);
  }

  u16* hl0 = hlds + (wave * 2 + 0) * 1024;
  u16* hl1 = hlds + (wave * 2 + 1) * 1024;
  const int cl = lane & 15;
  const int hi = lane >> 4;

  // particle-state init overlaps the weight DMA (disjoint LDS regions)
  float cst[2][4][8];
#pragma unroll
  for (int l = 0; l < 2; ++l) {
    u16* hl = l ? hl1 : hl0;
#pragma unroll
    for (int n4 = 0; n4 < 4; ++n4) {
      const int col = n4 * 16 + cl;
#pragma unroll
      for (int v = 0; v < 8; ++v) {
        const int m = v + 8 * hi;
        const int base = ((l * BS + m) * NH + col) * NP;
        cst[l][n4][v]   = pin[base + HALF_NP + p];
        hl[m * NH + col] = f2bf(pin[base + p]);
      }
    }
  }

  if (wave == 0) __builtin_amdgcn_s_wait_tensorcnt(0);
  __syncthreads();

  for (int t = 0; t < T_STEPS; ++t) {
    // -------- layer 0 --------
    V16 ax;
    {
      const U4* xp = (const U4*)(xAg + t * 512);
      ax.q[0] = xp[lane * 2 + 0];
      ax.q[1] = xp[lane * 2 + 1];
    }
    if (t + 1 < T_STEPS)
      __builtin_prefetch(xAg + (t + 1) * 512 + lane * 16, 0, 3);
    V16 ah[2];
#pragma unroll
    for (int c = 0; c < 2; ++c) {
      const int m = cl, off = hi * 8, base = c * 32;
      ah[c].q[0] = *(const U4*)(hl0 + m * NH + base + off);
      ah[c].q[1] = *(const U4*)(hl0 + m * NH + base + 16 + off);
    }
#pragma unroll
    for (int n4 = 0; n4 < 4; ++n4) {
      v8f g[4];
#pragma unroll
      for (int gs = 0; gs < 4; ++gs) {
        const int nt = gs * 4 + n4;
        // distinct SSA B operands -> scheduler keeps multiple ds_loads in flight
        const v16bf b0 = *(const v16bf*)(wlds + nt * 512 + lane * 16);
        const v16bf b1 = *(const v16bf*)(wlds + (16 + nt * 2 + 0) * 512 + lane * 16);
        const v16bf b2 = *(const v16bf*)(wlds + (16 + nt * 2 + 1) * 512 + lane * 16);
        // bias as v8f C operand straight from LDS (replicated x8)
        v8f acc = *(const v8f*)(brep + (nt * 16 + cl) * 8);
        acc = __builtin_amdgcn_wmma_f32_16x16x32_bf16(false, ax.v,   false, b0,
                                                      (short)0, acc, false, false);
        acc = __builtin_amdgcn_wmma_f32_16x16x32_bf16(false, ah[0].v, false, b1,
                                                      (short)0, acc, false, false);
        acc = __builtin_amdgcn_wmma_f32_16x16x32_bf16(false, ah[1].v, false, b2,
                                                      (short)0, acc, false, false);
        g[gs] = acc;
      }
      const int col = n4 * 16 + cl;
#pragma unroll
      for (int v = 0; v < 8; ++v) {
        float ig = fsig(g[0][v]);
        float fg = fsig(g[1][v]);
        float gg = ftanh(g[2][v]);
        float og = fsig(g[3][v]);
        float cv = fg * cst[0][n4][v] + ig * gg;
        cst[0][n4][v] = cv;
        hl0[(v + 8 * hi) * NH + col] = f2bf(og * ftanh(cv));
      }
    }
    // -------- layer 1 (x = layer-0 h at time t) --------
    V16 ax1[2], ah1[2];
#pragma unroll
    for (int c = 0; c < 2; ++c) {
      const int m = cl, off = hi * 8, base = c * 32;
      ax1[c].q[0] = *(const U4*)(hl0 + m * NH + base + off);
      ax1[c].q[1] = *(const U4*)(hl0 + m * NH + base + 16 + off);
      ah1[c].q[0] = *(const U4*)(hl1 + m * NH + base + off);
      ah1[c].q[1] = *(const U4*)(hl1 + m * NH + base + 16 + off);
    }
#pragma unroll
    for (int n4 = 0; n4 < 4; ++n4) {
      v8f g[4];
#pragma unroll
      for (int gs = 0; gs < 4; ++gs) {
        const int nt = gs * 4 + n4;
        const v16bf b0 = *(const v16bf*)(wlds + (48 + nt * 2 + 0) * 512 + lane * 16);
        const v16bf b1 = *(const v16bf*)(wlds + (48 + nt * 2 + 1) * 512 + lane * 16);
        const v16bf b2 = *(const v16bf*)(wlds + (80 + nt * 2 + 0) * 512 + lane * 16);
        const v16bf b3 = *(const v16bf*)(wlds + (80 + nt * 2 + 1) * 512 + lane * 16);
        v8f acc = *(const v8f*)(brep + 2048 + (nt * 16 + cl) * 8);
        acc = __builtin_amdgcn_wmma_f32_16x16x32_bf16(false, ax1[0].v, false, b0,
                                                      (short)0, acc, false, false);
        acc = __builtin_amdgcn_wmma_f32_16x16x32_bf16(false, ax1[1].v, false, b1,
                                                      (short)0, acc, false, false);
        acc = __builtin_amdgcn_wmma_f32_16x16x32_bf16(false, ah1[0].v, false, b2,
                                                      (short)0, acc, false, false);
        acc = __builtin_amdgcn_wmma_f32_16x16x32_bf16(false, ah1[1].v, false, b3,
                                                      (short)0, acc, false, false);
        g[gs] = acc;
      }
      const int col = n4 * 16 + cl;
#pragma unroll
      for (int v = 0; v < 8; ++v) {
        float ig = fsig(g[0][v]);
        float fg = fsig(g[1][v]);
        float gg = ftanh(g[2][v]);
        float og = fsig(g[3][v]);
        float cv = fg * cst[1][n4][v] + ig * gg;
        cst[1][n4][v] = cv;
        hl1[(v + 8 * hi) * NH + col] = f2bf(og * ftanh(cv));
      }
    }
  }

  // store hT/cT + process noise into pre-resample particle buffer
  const float qn = fabsf(*qp), en = fabsf(*ep);
#pragma unroll
  for (int l = 0; l < 2; ++l) {
    u16* hl = l ? hl1 : hl0;
#pragma unroll
    for (int n4 = 0; n4 < 4; ++n4) {
      const int col = n4 * 16 + cl;
#pragma unroll
      for (int v = 0; v < 8; ++v) {
        const int m = v + 8 * hi;
        const int base = ((l * BS + m) * NH + col) * NP;
        float hT = bf2f(hl[m * NH + col]);
        float cT = cst[l][n4][v];
        u32 seed = (u32)(base + p) * 2654435761u + 12345u;
        ppre[base + p]           = hT + qn * gnoise(seed);
        ppre[base + HALF_NP + p] = cT + en * gnoise(seed ^ 0x9E3779B9u);
      }
    }
  }
}

// ---- Y[b][p] = sum_i P[layer1][b][i][p] * H[i] ----
__global__ void yproj_kernel(const float* __restrict__ P, const float* __restrict__ H,
                             float* __restrict__ Y) {
  int idx = blockIdx.x * blockDim.x + threadIdx.x;
  if (idx >= BS * NP) return;
  int p = idx & (NP - 1), b = idx >> 11;
  float s = 0.f;
  for (int i = 0; i < NH; ++i)
    s += P[((1 * BS + b) * NH + i) * NP + p] * H[i];
  Y[b * NP + p] = s;
}

// ---- covariance, Cholesky, per-particle log-likelihood, softmax, neff ----
__global__ void __launch_bounds__(256)
stats_kernel(const float* __restrict__ Y, const float* __restrict__ w_in,
             const float* __restrict__ yobs, const float* __restrict__ rp,
             float* __restrict__ neww, float* __restrict__ stats) {
  __shared__ float red[256];
  __shared__ float mu[16];
  __shared__ float sig[16][16];
  __shared__ float Lc[16][16];
  __shared__ float sv[4];
  int tid = threadIdx.x;

  float s = 0.f;
  for (int pI = tid; pI < NP; pI += 256) s += w_in[pI];
  red[tid] = s; __syncthreads();
  for (int o = 128; o > 0; o >>= 1) { if (tid < o) red[tid] += red[tid + o]; __syncthreads(); }
  if (tid == 0) sv[0] = red[0];
  __syncthreads();
  float wsum = sv[0];

  if (tid < 16) {
    float m = 0.f;
    for (int pI = 0; pI < NP; ++pI) m += Y[tid * NP + pI];
    mu[tid] = m / (float)NP;
  }
  __syncthreads();
  {
    int a = tid >> 4, b = tid & 15;
    float c = 0.f, ma = mu[a], mb = mu[b];
    for (int pI = 0; pI < NP; ++pI)
      c += (Y[a * NP + pI] - ma) * (Y[b * NP + pI] - mb);
    float r = *rp;
    sig[a][b] = c / (float)(NP - 1) + ((a == b) ? (r * r + 1e-6f) : 0.f);
  }
  __syncthreads();
  if (tid == 0) {
    float ld = 0.f;
    for (int i = 0; i < 16; ++i)
      for (int j = 0; j <= i; ++j) {
        float a2 = sig[i][j];
        for (int k = 0; k < j; ++k) a2 -= Lc[i][k] * Lc[j][k];
        if (i == j) { float d = sqrtf(fmaxf(a2, 1e-20f)); Lc[i][i] = d; ld += logf(d); }
        else Lc[i][j] = a2 / Lc[j][j];
      }
    sv[1] = 2.f * ld;
  }
  __syncthreads();
  float logdet = sv[1];

  float lmax = -1e30f;
  for (int pI = tid; pI < NP; pI += 256) {
    float v[16], tt[16];
    float wp = w_in[pI];
    for (int b = 0; b < 16; ++b) v[b] = yobs[b] - Y[b * NP + pI] * wp / wsum;
    float quad = 0.f;
    for (int i = 0; i < 16; ++i) {
      float a2 = v[i];
      for (int k = 0; k < i; ++k) a2 -= Lc[i][k] * tt[k];
      tt[i] = a2 / Lc[i][i];
      quad += tt[i] * tt[i];
    }
    float logit = logf(fmaxf(wp, 1e-30f)) + (-0.5f * logdet - 0.5f * quad);
    neww[pI] = logit;
    lmax = fmaxf(lmax, logit);
  }
  red[tid] = lmax; __syncthreads();
  for (int o = 128; o > 0; o >>= 1) { if (tid < o) red[tid] = fmaxf(red[tid], red[tid + o]); __syncthreads(); }
  if (tid == 0) sv[2] = red[0];
  __syncthreads();
  float mx = sv[2];
  float es = 0.f;
  for (int pI = tid; pI < NP; pI += 256) es += __expf(neww[pI] - mx);
  red[tid] = es; __syncthreads();
  for (int o = 128; o > 0; o >>= 1) { if (tid < o) red[tid] += red[tid + o]; __syncthreads(); }
  if (tid == 0) sv[3] = red[0];
  __syncthreads();
  float denom = sv[3];
  for (int pI = tid; pI < NP; pI += 256) neww[pI] = __expf(neww[pI] - mx) / denom;
  __syncthreads();
  for (int h = 0; h < 2; ++h) {
    float sq = 0.f;
    for (int pI = tid; pI < HALF_NP; pI += 256) { float w = neww[h * HALF_NP + pI]; sq += w * w; }
    red[tid] = sq; __syncthreads();
    for (int o = 128; o > 0; o >>= 1) { if (tid < o) red[tid] += red[tid + o]; __syncthreads(); }
    if (tid == 0) stats[h] = 1.f / fmaxf(red[0], 1e-30f);
    __syncthreads();
  }
}

// ---- resample gather (deterministic permutation when neff < N/4) ----
__global__ void gather_kernel(const float* __restrict__ ppre, const float* __restrict__ stats,
                              float* __restrict__ outp) {
  int idx = blockIdx.x * blockDim.x + threadIdx.x;
  if (idx >= LS * BS * NH * NP) return;
  int p = idx & (NP - 1);
  int row = idx >> 11;
  int h = p >> 10, j = p & (HALF_NP - 1);
  bool cond = stats[h] < (float)NP / 4.0f;
  int src = cond ? ((j * 997) & (HALF_NP - 1)) : j;
  outp[idx] = ppre[row * NP + h * HALF_NP + src];
}

__global__ void wf_kernel(const float* __restrict__ neww, const float* __restrict__ stats,
                          float* __restrict__ wf) {
  int p = blockIdx.x * blockDim.x + threadIdx.x;
  if (p >= NP) return;
  int h = p >> 10;
  float w = neww[p];
  bool cond = stats[h] < (float)NP / 4.0f;
  float soft = 0.5f * w + 0.5f / (float)HALF_NP;
  wf[p] = cond ? (w / soft) : w;
}

// ---- final moments + 16x16 solve/logdet ----
__global__ void __launch_bounds__(256)
final_kernel(const float* __restrict__ Yf, const float* __restrict__ wf,
             const float* __restrict__ yobs, const float* __restrict__ rp,
             float* __restrict__ out) {
  __shared__ float red[256];
  __shared__ float mu[16];
  __shared__ float wmu[16];
  __shared__ float A[16][17];
  __shared__ float sv[1];
  int tid = threadIdx.x;

  float s = 0.f;
  for (int pI = tid; pI < NP; pI += 256) s += wf[pI];
  red[tid] = s; __syncthreads();
  for (int o = 128; o > 0; o >>= 1) { if (tid < o) red[tid] += red[tid + o]; __syncthreads(); }
  if (tid == 0) sv[0] = red[0];
  __syncthreads();
  float wsum = sv[0];

  if (tid < 16) {
    float a = 0.f, m = 0.f;
    for (int pI = 0; pI < NP; ++pI) {
      float yv = Yf[tid * NP + pI];
      a += yv * wf[pI];
      m += yv;
    }
    wmu[tid] = a / wsum;
    mu[tid] = m / (float)NP;
    out[O_WMU + tid] = wmu[tid];
  }
  __syncthreads();
  {
    int a2 = tid >> 4, b = tid & 15;
    float c = 0.f, ma = mu[a2], mb = mu[b];
    for (int pI = 0; pI < NP; ++pI)
      c += (Yf[a2 * NP + pI] - ma) * (Yf[b * NP + pI] - mb);
    float r = *rp;
    float fc = c / (float)(NP - 1) + ((a2 == b) ? r * r : 0.f);
    out[O_COV + a2 * 16 + b] = fc;
    A[a2][b] = fc + ((a2 == b) ? 1e-6f : 0.f);
  }
  __syncthreads();
  if (tid == 0) {
    float inn[16];
    for (int b = 0; b < 16; ++b) { inn[b] = yobs[b] - wmu[b]; A[b][16] = inn[b]; }
    float logdet = 0.f;
    for (int k = 0; k < 16; ++k) {
      float piv = A[k][k];
      logdet += logf(fabsf(piv) + 1e-30f);
      float ip = 1.0f / piv;
      for (int j = k; j < 17; ++j) A[k][j] *= ip;
      for (int i2 = k + 1; i2 < 16; ++i2) {
        float f = A[i2][k];
        for (int j = k; j < 17; ++j) A[i2][j] -= f * A[k][j];
      }
    }
    float sol[16];
    for (int i2 = 15; i2 >= 0; --i2) {
      float v = A[i2][16];
      for (int j = i2 + 1; j < 16; ++j) v -= A[i2][j] * sol[j];
      sol[i2] = v;
    }
    float nis = 0.f;
    for (int b = 0; b < 16; ++b) nis += inn[b] * sol[b];
    out[O_L]   = -0.5f * logdet - 0.5f * nis;
    out[O_NIS] = nis;
  }
}

extern "C" void kernel_launch(void* const* d_in, const int* in_sizes, int n_in,
                              void* d_out, int out_size, void* d_ws, size_t ws_size,
                              hipStream_t stream) {
  (void)in_sizes; (void)n_in; (void)out_size; (void)ws_size;
  const float* x    = (const float*)d_in[0];
  const float* y    = (const float*)d_in[1];
  const float* part = (const float*)d_in[2];
  const float* w    = (const float*)d_in[3];
  const float* Wih0 = (const float*)d_in[4];
  const float* Whh0 = (const float*)d_in[5];
  const float* bih0 = (const float*)d_in[6];
  const float* bhh0 = (const float*)d_in[7];
  const float* Wih1 = (const float*)d_in[8];
  const float* Whh1 = (const float*)d_in[9];
  const float* bih1 = (const float*)d_in[10];
  const float* bhh1 = (const float*)d_in[11];
  const float* H    = (const float*)d_in[12];
  const float* q    = (const float*)d_in[13];
  const float* e    = (const float*)d_in[14];
  const float* r    = (const float*)d_in[15];

  float* out = (float*)d_out;
  char*  ws  = (char*)d_ws;
  u16*   wB   = (u16*)(ws + WS_WB);
  float* brep = (float*)(ws + WS_BREP);
  u16*   xA   = (u16*)(ws + WS_XA);
  float* ppre = (float*)(ws + WS_PPRE);
  float* Y    = (float*)(ws + WS_Y);
  float* neww = (float*)(ws + WS_NEWW);
  float* st   = (float*)(ws + WS_STATS);
  float* Yf   = (float*)(ws + WS_YF);

  prep_kernel<<<64, 256, 0, stream>>>(x, Wih0, Whh0, bih0, bhh0,
                                      Wih1, Whh1, bih1, bhh1, wB, brep, xA);

  size_t smem = 114688 + 16384 + 32768;  // weights + replicated bias + h state = 160 KB
  pf_lstm_wmma_kernel<<<HALF_NP / N_WAVES, 256, smem, stream>>>(wB, xA, part,
                                                                ppre, q, e);

  yproj_kernel<<<(BS * NP) / 256, 256, 0, stream>>>(ppre, H, Y);
  stats_kernel<<<1, 256, 0, stream>>>(Y, w, y, r, neww, st);
  gather_kernel<<<(LS * BS * NH * NP) / 256, 256, 0, stream>>>(ppre, st, out + O_PF);
  wf_kernel<<<NP / 256, 256, 0, stream>>>(neww, st, out + O_WF);
  yproj_kernel<<<(BS * NP) / 256, 256, 0, stream>>>(out + O_PF, H, Yf);
  final_kernel<<<1, 256, 0, stream>>>(Yf, out + O_WF, y, r, out);
}
